// MoE_layer_time_35313221108107
// MI455X (gfx1250) — matrix operand order repaired and checked by reading, not compiled, not run on hardware
//
#include <hip/hip_runtime.h>
#include <hip/hip_bf16.h>
#include <math.h>

// ---------------- problem constants ----------------
#define BATCH   8
#define CIN     36
#define HH      256
#define WW      256
#define NEXP    6
#define TDIM    32
#define HID     72        // CIN*2
#define HW      (HH * WW)
#define YELEMS  (BATCH * CIN * HW)

// ---------------- tile / padding constants ----------------
#define TH      16        // output tile height
#define TW      16        // output tile width
#define CINP    64        // input channels padded (36 -> 64, 2 K-steps of 32)
#define HIDP    96        // hidden channels padded (72 -> 96, 3 K-steps of 32)
#define YCHP    48        // output channels padded (36 -> 48)
#define MIDD    18        // mid tile dim (TH+2, halo for conv2)
#define IND     20        // input tile dim (TH+4)
#define NPOS1   (MIDD * MIDD)       // 324 mid positions
#define NT1     21        // ceil(324/16) N tiles for GEMM1
#define MT1     5         // 80/16 M tiles for GEMM1 (72 padded to 80)
#define MT2     3         // 48/16 M tiles for GEMM2 (36 padded to 48)
#define NT2     16        // 256/16 N tiles for GEMM2

#define XS_ELEMS  (IND * IND * CINP)     // 25600 f16
#define MS_ELEMS  (MIDD * MIDD * HIDP)   // 31104 f16
#define W1_ELEMS  (MT1 * 16 * 9 * CINP)  // 46080 f16 (one expert, packed+padded)
#define W2_ELEMS  (MT2 * 16 * 9 * HIDP)  // 41472 f16 (one expert, packed+padded)
#define WB_ELEMS  W1_ELEMS               // union buffer (max of the two)
#define YT_ELEMS  (TH * TW * YCHP)       // 12288 f32
#define LDS_BYTES ((XS_ELEMS + MS_ELEMS + WB_ELEMS) * 2 + YT_ELEMS * 4)

typedef _Float16 half_t;
typedef __attribute__((ext_vector_type(16))) _Float16 v16h;
typedef __attribute__((ext_vector_type(8)))  float    v8f;
typedef int vint4 __attribute__((__vector_size__(16)));

union F16x16 { v16h v; uint4 q[2]; };
union H8     { half_t h[8]; uint4 q; };

// ---- gfx1250 async global->LDS copy (guarded; falls back to plain copy) ----
#if defined(__gfx1250__) && \
    __has_builtin(__builtin_amdgcn_global_load_async_to_lds_b128) && \
    __has_builtin(__builtin_amdgcn_s_wait_asynccnt)
#define HAVE_ASYNC_LDS 1
typedef __attribute__((address_space(1))) vint4 gvint4_t;
typedef __attribute__((address_space(3))) vint4 lvint4_t;
#else
#define HAVE_ASYNC_LDS 0
#endif

// =====================================================================
// Kernel 1: per-(b,c) avg + max pool over 256x256
// =====================================================================
__global__ __launch_bounds__(256) void pool_kernel(const float* __restrict__ x,
                                                   float* __restrict__ pooled) {
  const int bc = blockIdx.x;                  // b*36 + c
  const float* p = x + (size_t)bc * HW;
  float s = 0.f, m = -3.4e38f;
  for (int i = threadIdx.x; i < HW; i += 256) {
    float v = p[i];
    s += v;
    m = fmaxf(m, v);
  }
  __shared__ float ss[256];
  __shared__ float sm[256];
  ss[threadIdx.x] = s;
  sm[threadIdx.x] = m;
  __syncthreads();
  for (int o = 128; o > 0; o >>= 1) {
    if (threadIdx.x < o) {
      ss[threadIdx.x] += ss[threadIdx.x + o];
      sm[threadIdx.x] = fmaxf(sm[threadIdx.x], sm[threadIdx.x + o]);
    }
    __syncthreads();
  }
  if (threadIdx.x == 0) {
    int b = bc / CIN, c = bc % CIN;
    pooled[b * HID + c]       = ss[0] * (1.0f / (float)HW);  // avg half
    pooled[b * HID + CIN + c] = sm[0];                       // max half
  }
}

// =====================================================================
// Kernel 2: gate MLP + top-2 softmax + CV-squared loss
// =====================================================================
__device__ inline float cvsq6(const float* v) {
  float mean = 0.f;
  for (int i = 0; i < 6; i++) mean += v[i];
  mean *= (1.f / 6.f);
  float var = 0.f;
  for (int i = 0; i < 6; i++) { float d = v[i] - mean; var += d * d; }
  var *= (1.f / 5.f);                         // unbiased (ddof=1)
  return var / (mean * mean + 1e-10f);
}

__global__ __launch_bounds__(256) void gate_kernel(
    const float* __restrict__ pooled, const float* __restrict__ timev,
    const float* __restrict__ time_w, const float* __restrict__ time_b,
    const float* __restrict__ gw1, const float* __restrict__ gb1,
    const float* __restrict__ gw2, const float* __restrict__ gb2,
    float* __restrict__ gates, float* __restrict__ loss_out) {
  __shared__ float vbuf[BATCH * HID];
  __shared__ float hbuf[BATCH * HID];
  __shared__ float lg[BATCH * NEXP];
  __shared__ float gl[BATCH * NEXP];
  const int tid = threadIdx.x;

  for (int i = tid; i < BATCH * HID; i += 256) {
    int b = i / HID, j = i % HID;
    float a = pooled[i] + time_b[j];
    for (int t = 0; t < TDIM; t++) a += timev[b * TDIM + t] * time_w[t * HID + j];
    vbuf[i] = a;
  }
  __syncthreads();
  for (int i = tid; i < BATCH * HID; i += 256) {
    int b = i / HID, j = i % HID;
    float a = gb1[j];
    for (int k = 0; k < HID; k++) a += vbuf[b * HID + k] * gw1[k * HID + j];
    hbuf[i] = a > 0.f ? a : 0.01f * a;        // LeakyReLU(0.01)
  }
  __syncthreads();
  for (int i = tid; i < BATCH * NEXP; i += 256) {
    int b = i / NEXP, j = i % NEXP;
    float a = gb2[j];
    for (int k = 0; k < HID; k++) a += hbuf[b * HID + k] * gw2[k * NEXP + j];
    lg[i] = a;
  }
  __syncthreads();
  if (tid < BATCH) {
    int b = tid;
    int i0 = 0; float v0 = lg[b * NEXP];
    for (int j = 1; j < NEXP; j++) if (lg[b * NEXP + j] > v0) { v0 = lg[b * NEXP + j]; i0 = j; }
    int i1 = -1; float v1 = -3.4e38f;
    for (int j = 0; j < NEXP; j++)
      if (j != i0 && lg[b * NEXP + j] > v1) { v1 = lg[b * NEXP + j]; i1 = j; }
    float e0 = 1.f, e1 = expf(v1 - v0);
    float inv = 1.f / (e0 + e1);
    for (int j = 0; j < NEXP; j++) gl[b * NEXP + j] = 0.f;
    gl[b * NEXP + i0] = e0 * inv;
    gl[b * NEXP + i1] = e1 * inv;
    for (int j = 0; j < NEXP; j++) gates[b * NEXP + j] = gl[b * NEXP + j];
  }
  __syncthreads();
  if (tid == 0) {
    float imp[NEXP], ld[NEXP];
    for (int e = 0; e < NEXP; e++) {
      float si = 0.f, sl = 0.f;
      for (int b = 0; b < BATCH; b++) {
        float g = gl[b * NEXP + e];
        si += g;
        sl += (g > 0.f) ? 1.f : 0.f;
      }
      imp[e] = si; ld[e] = sl;
    }
    loss_out[0] = (cvsq6(imp) + cvsq6(ld)) * 0.01f;
  }
}

// =====================================================================
// Kernel 2b: one-time weight pack: f32 OIHW -> f16, padded, LDS layout
//   w1p: [E][80 oc][9 tap][64 ic]   w2p: [E][48 oc][9 tap][96 ic]
// =====================================================================
__global__ __launch_bounds__(256) void pack_weights_kernel(
    const float* __restrict__ w1, const float* __restrict__ w2,
    half_t* __restrict__ w1p, half_t* __restrict__ w2p) {
  int i = blockIdx.x * 256 + threadIdx.x;
  if (i < NEXP * W1_ELEMS) {
    int e = i / W1_ELEMS, r = i % W1_ELEMS;
    int ic = r % CINP;
    int q  = r / CINP;
    int tap = q % 9, oc = q / 9;
    float v = 0.f;
    if (oc < HID && ic < CIN)
      v = w1[(((size_t)e * HID + oc) * CIN + ic) * 9 + tap];
    w1p[i] = (half_t)v;
  }
  if (i < NEXP * W2_ELEMS) {
    int e = i / W2_ELEMS, r = i % W2_ELEMS;
    int ic = r % HIDP;
    int q  = r / HIDP;
    int tap = q % 9, oc = q / 9;
    float v = 0.f;
    if (oc < CIN && ic < HID)
      v = w2[(((size_t)e * CIN + oc) * HID + ic) * 9 + tap];
    w2p[i] = (half_t)v;
  }
}

// ---- fill LDS buffer from packed f16 global (async path if available) ----
__device__ inline void fill_lds_weights(half_t* dst, const half_t* src,
                                        int n_halfs, int tid) {
#if HAVE_ASYNC_LDS
  for (int i = tid; i < n_halfs / 8; i += 256) {
    __builtin_amdgcn_global_load_async_to_lds_b128(
        (gvint4_t*)(vint4*)(src + i * 8),
        (lvint4_t*)(vint4*)(dst + i * 8), 0, 0);
  }
  __builtin_amdgcn_s_wait_asynccnt(0);
#else
  for (int i = tid; i < n_halfs / 8; i += 256)
    *(uint4*)(dst + i * 8) = *(const uint4*)(src + i * 8);
#endif
}

// =====================================================================
// Kernel 3: fused MoE experts — implicit-GEMM conv3x3 via WMMA f16
//   block = 16x16 spatial tile of one image, 8 wave32
//   per K-step: load B frag + ALL M-tile A frags first, then issue the
//   WMMA chain back-to-back (single dscnt wait per step, overlappable)
// =====================================================================
__global__ __launch_bounds__(256, 1) void moe_conv_kernel(
    const float* __restrict__ x, const half_t* __restrict__ w1p,
    const half_t* __restrict__ w2p, const float* __restrict__ gates,
    float* __restrict__ y) {
  extern __shared__ char smemraw[];
  half_t* xs    = (half_t*)smemraw;           // [IND][IND][CINP]
  half_t* ms    = xs + XS_ELEMS;              // [MIDD][MIDD][HIDP]
  half_t* wbuf  = ms + MS_ELEMS;              // w1 or w2 (union)
  float*  ytile = (float*)(wbuf + WB_ELEMS);  // [256 pos][YCHP] accumulator

  const int b   = blockIdx.z;
  const int gy0 = blockIdx.y * TH;
  const int gx0 = blockIdx.x * TW;
  const int tid  = threadIdx.x;
  const int lane = tid & 31;
  const int wave = tid >> 5;
  const int Lm   = lane & 15;
  const int hi   = lane >> 4;

  // ---- fill padded input halo tile (f32 -> f16, zero pad) ----
  for (int i = tid; i < XS_ELEMS; i += 256) {
    int ic = i % CINP;
    int r  = i / CINP;
    int ix = r % IND, iy = r / IND;
    int gy = gy0 - 2 + iy, gx = gx0 - 2 + ix;
    float v = 0.f;
    if (ic < CIN && gy >= 0 && gy < HH && gx >= 0 && gx < WW)
      v = x[(((size_t)b * CIN + ic) * HH + gy) * WW + gx];
    xs[i] = (half_t)v;
  }
  // ---- zero mid tile (K padding for GEMM2 must be clean zeros) ----
  for (int i = tid; i < MS_ELEMS; i += 256) ms[i] = (half_t)0.f;
  // ---- zero y accumulator tile ----
  for (int i = tid; i < YT_ELEMS; i += 256) ytile[i] = 0.f;

  for (int e = 0; e < NEXP; e++) {
    // prefetch next expert's packed weights toward L2
    {
      int en = (e + 1) % NEXP;
      const half_t* nw1 = w1p + (size_t)en * W1_ELEMS;
      const half_t* nw2 = w2p + (size_t)en * W2_ELEMS;
      for (int i = tid * 64; i < W1_ELEMS; i += 256 * 64)
        __builtin_prefetch(nw1 + i, 0, 1);
      for (int i = tid * 64; i < W2_ELEMS; i += 256 * 64)
        __builtin_prefetch(nw2 + i, 0, 1);
    }
    float g = gates[b * NEXP + e];
    __syncthreads();                          // wbuf/ms free for reuse
    if (g == 0.f) continue;                   // block-uniform: b is per-block

    // ---- conv1 weights -> LDS (pure f16 copy, async if available) ----
    fill_lds_weights(wbuf, w1p + (size_t)e * W1_ELEMS, W1_ELEMS, tid);
    __syncthreads();

    // ---- GEMM1: mid[72 x 324] = W1 * im2col(x), + GELU ----
    for (int nt = wave; nt < NT1; nt += 8) {
      int n0 = nt * 16;
      int n  = n0 + Lm;
      int nn = n < NPOS1 ? n : NPOS1 - 1;     // clamp pad columns in-bounds
      int my = nn / MIDD, mx = nn % MIDD;
      v8f acc[MT1];
#pragma unroll
      for (int m = 0; m < MT1; m++)
#pragma unroll
        for (int r = 0; r < 8; r++) acc[m][r] = 0.f;

#pragma unroll 3
      for (int tap = 0; tap < 9; tap++) {
        int dy = tap / 3, dx = tap % 3;
        const half_t* bp = xs + ((my + dy) * IND + (mx + dx)) * CINP + hi * 8;
        const half_t* ap0 = wbuf + (Lm * 9 + tap) * CINP + hi * 8;
#pragma unroll
        for (int k0 = 0; k0 < CINP; k0 += 32) {
          // load B fragment and ALL A fragments up-front
          F16x16 bf, af[MT1];
          bf.q[0] = *(const uint4*)(bp + k0);
          bf.q[1] = *(const uint4*)(bp + k0 + 16);
#pragma unroll
          for (int m = 0; m < MT1; m++) {
            const half_t* ap = ap0 + m * (16 * 9 * CINP) + k0;
            af[m].q[0] = *(const uint4*)(ap);
            af[m].q[1] = *(const uint4*)(ap + 16);
          }
          // WMMA chain back-to-back
#pragma unroll
          for (int m = 0; m < MT1; m++)
            acc[m] = __builtin_amdgcn_wmma_f32_16x16x32_f16(
                false, af[m].v, false, bf.v, (short)0, acc[m], false, false);
        }
      }
      if (n < NPOS1) {                        // drop pad columns
#pragma unroll
        for (int m = 0; m < MT1; m++) {
          H8 o;
#pragma unroll
          for (int r = 0; r < 8; r++) {
            float v = acc[m][r];
            v = 0.5f * v * (1.f + erff(v * 0.70710678118654752f));  // exact GELU
            o.h[r] = (half_t)v;
          }
          *(uint4*)(ms + n * HIDP + m * 16 + hi * 8) = o.q;  // ds_store_b128
        }
      }
    }
    __syncthreads();

    // ---- conv2 weights -> LDS ----
    fill_lds_weights(wbuf, w2p + (size_t)e * W2_ELEMS, W2_ELEMS, tid);
    __syncthreads();

    // ---- GEMM2: y_e[36 x 256] = W2 * im2col(mid); ytile += g * y_e ----
    for (int nt = wave; nt < NT2; nt += 8) {
      int n0 = nt * 16;
      int n  = n0 + Lm;
      int py = n >> 4, px = n & 15;
      v8f acc[MT2];
#pragma unroll
      for (int m = 0; m < MT2; m++)
#pragma unroll
        for (int r = 0; r < 8; r++) acc[m][r] = 0.f;

#pragma unroll 3
      for (int tap = 0; tap < 9; tap++) {
        int dy = tap / 3, dx = tap % 3;
        const half_t* bp = ms + ((py + dy) * MIDD + (px + dx)) * HIDP + hi * 8;
        const half_t* ap0 = wbuf + (Lm * 9 + tap) * HIDP + hi * 8;
#pragma unroll
        for (int k0 = 0; k0 < HIDP; k0 += 32) {
          F16x16 bf, af[MT2];
          bf.q[0] = *(const uint4*)(bp + k0);
          bf.q[1] = *(const uint4*)(bp + k0 + 16);
#pragma unroll
          for (int m = 0; m < MT2; m++) {
            const half_t* ap = ap0 + m * (16 * 9 * HIDP) + k0;
            af[m].q[0] = *(const uint4*)(ap);
            af[m].q[1] = *(const uint4*)(ap + 16);
          }
#pragma unroll
          for (int m = 0; m < MT2; m++)
            acc[m] = __builtin_amdgcn_wmma_f32_16x16x32_f16(
                false, af[m].v, false, bf.v, (short)0, acc[m], false, false);
        }
      }
      // gate-weighted accumulate into LDS ytile (unique owner per element)
#pragma unroll
      for (int m = 0; m < MT2; m++) {
        float* yp = ytile + n * YCHP + m * 16 + hi * 8;
#pragma unroll
        for (int r = 0; r < 8; r++) yp[r] += g * acc[m][r];
      }
    }
  }
  __syncthreads();

  // ---- store y tile to global ----
  for (int i = tid; i < CIN * (TH * TW); i += 256) {
    int ch = i >> 8;            // 0..35
    int p  = i & 255;           // position in tile
    int py = p >> 4, px = p & 15;
    y[(((size_t)b * CIN + ch) * HH + (gy0 + py)) * WW + (gx0 + px)] =
        ytile[p * YCHP + ch];
  }
}

// =====================================================================
extern "C" void kernel_launch(void* const* d_in, const int* in_sizes, int n_in,
                              void* d_out, int out_size, void* d_ws, size_t ws_size,
                              hipStream_t stream) {
  const float* x      = (const float*)d_in[0];
  const float* timev  = (const float*)d_in[1];
  const float* time_w = (const float*)d_in[2];
  const float* time_b = (const float*)d_in[3];
  const float* gw1    = (const float*)d_in[4];
  const float* gb1    = (const float*)d_in[5];
  const float* gw2    = (const float*)d_in[6];
  const float* gb2    = (const float*)d_in[7];
  const float* w1     = (const float*)d_in[8];
  const float* w2     = (const float*)d_in[9];

  float* y        = (float*)d_out;
  float* loss_out = y + YELEMS;               // tuple output: (y, loss)
  float* pooled   = (float*)d_ws;             // [B][2C] = 576 floats
  float* gates    = pooled + BATCH * HID;     // [B][E]  = 48 floats
  half_t* w1p     = (half_t*)((char*)d_ws + 4096);        // packed f16 weights
  half_t* w2p     = w1p + (size_t)NEXP * W1_ELEMS;        // (total ~1.05 MB)

  pool_kernel<<<BATCH * CIN, 256, 0, stream>>>(x, pooled);
  gate_kernel<<<1, 256, 0, stream>>>(pooled, timev, time_w, time_b,
                                     gw1, gb1, gw2, gb2, gates, loss_out);
  {
    int npack = NEXP * (W1_ELEMS > W2_ELEMS ? W1_ELEMS : W2_ELEMS);
    pack_weights_kernel<<<(npack + 255) / 256, 256, 0, stream>>>(w1, w2, w1p, w2p);
  }

  (void)hipFuncSetAttribute((const void*)moe_conv_kernel,
                            hipFuncAttributeMaxDynamicSharedMemorySize,
                            (int)LDS_BYTES);
  moe_conv_kernel<<<dim3(WW / TW, HH / TH, BATCH), 256, LDS_BYTES, stream>>>(
      x, w1p, w2p, gates, y);
}
